// MoELayer_56521769616154
// MI455X (gfx1250) — matrix operand order repaired and checked by reading, not compiled, not run on hardware
//
#include <hip/hip_runtime.h>
#include <hip/hip_bf16.h>

#define N_TOK   4096
#define HID     1024
#define EXP_DIM 4096
#define NEXP    8

typedef __attribute__((ext_vector_type(16))) __bf16        v16bf;
typedef __attribute__((ext_vector_type(8)))  float         v8f;
typedef __attribute__((ext_vector_type(4)))  unsigned int  v4u;

union FragU { v4u u[2]; v16bf v; };

__device__ __forceinline__ unsigned short f2bf(float f) {
    unsigned int u = __float_as_uint(f);
    unsigned int r = (u + 0x7FFFu + ((u >> 16) & 1u)) >> 16;   // RNE
    return (unsigned short)r;
}

__device__ __forceinline__ v8f zero8() {
    v8f z = {0.f,0.f,0.f,0.f,0.f,0.f,0.f,0.f};
    return z;
}

// ---------------- pack kernels ----------------

__global__ void pack_x_kernel(const float* __restrict__ x,
                              unsigned short* __restrict__ xb, int n) {
    int i = blockIdx.x * 256 + threadIdx.x;
    if (i < n) xb[i] = f2bf(x[i]);
}

// per expert: src is [R][C] fp32, dst is [C][R] bf16 (column-panel layout)
__global__ void pack_transpose_kernel(const float* __restrict__ src,
                                      unsigned short* __restrict__ dst,
                                      int R, int C) {
    __shared__ float t[32][33];
    const int e = blockIdx.z;
    const float* s = src + (size_t)e * R * C;
    unsigned short* d = dst + (size_t)e * R * C;
    const int c0 = blockIdx.x * 32;
    const int r0 = blockIdx.y * 32;
    for (int i = threadIdx.y; i < 32; i += 8)
        t[i][threadIdx.x] = s[(size_t)(r0 + i) * C + c0 + threadIdx.x];
    __syncthreads();
    for (int i = threadIdx.y; i < 32; i += 8)
        d[(size_t)(c0 + i) * R + r0 + threadIdx.x] = f2bf(t[threadIdx.x][i]);
}

// ---------------- gating ----------------

__global__ __launch_bounds__(256)
void gate_kernel(const float* __restrict__ x, const float* __restrict__ Wg,
                 const float* __restrict__ bg, float* __restrict__ probs,
                 int* __restrict__ cnt, int* __restrict__ list,
                 float* __restrict__ wtop) {
    const int lane = threadIdx.x & 31;
    const int wv   = threadIdx.x >> 5;
    const int tok  = blockIdx.x * 8 + wv;
    if (tok >= N_TOK) return;

    float p[NEXP];
#pragma unroll
    for (int e = 0; e < NEXP; ++e) p[e] = 0.f;

    for (int h = lane; h < HID; h += 32) {
        float xv = x[(size_t)tok * HID + h];
        const float* wr = Wg + (size_t)h * NEXP;
#pragma unroll
        for (int e = 0; e < NEXP; ++e) p[e] += xv * wr[e];
    }
#pragma unroll
    for (int off = 16; off; off >>= 1) {
#pragma unroll
        for (int e = 0; e < NEXP; ++e) p[e] += __shfl_xor(p[e], off, 32);
    }

    if (lane == 0) {
        float lg[NEXP];
        float mx = -1e30f;
#pragma unroll
        for (int e = 0; e < NEXP; ++e) { lg[e] = p[e] + bg[e]; mx = lg[e] > mx ? lg[e] : mx; }
        float sum = 0.f;
#pragma unroll
        for (int e = 0; e < NEXP; ++e) { lg[e] = __expf(lg[e] - mx); sum += lg[e]; }
        float inv = 1.f / sum;
#pragma unroll
        for (int e = 0; e < NEXP; ++e) {
            lg[e] *= inv;
            probs[(size_t)tok * NEXP + e] = lg[e];
        }
        int i0 = 0;
#pragma unroll
        for (int e = 1; e < NEXP; ++e) if (lg[e] > lg[i0]) i0 = e;
        int i1 = (i0 == 0) ? 1 : 0;
#pragma unroll
        for (int e = 0; e < NEXP; ++e)
            if (e != i0 && lg[e] > lg[i1]) i1 = e;
        float g0 = lg[i0], g1 = lg[i1];
        float gs = 1.f / (g0 + g1);
        wtop[tok * 2 + 0] = g0 * gs;
        wtop[tok * 2 + 1] = g1 * gs;
        int s0 = atomicAdd(&cnt[i0], 1);
        list[i0 * N_TOK + s0] = tok * 2 + 0;
        int s1 = atomicAdd(&cnt[i1], 1);
        list[i1 * N_TOK + s1] = tok * 2 + 1;
    }
}

// deterministic per-expert prob sums
__global__ void counts_kernel(const float* __restrict__ probs,
                              float* __restrict__ counts) {
    __shared__ float red[256];
    const int e = blockIdx.x;
    float s = 0.f;
    for (int n = threadIdx.x; n < N_TOK; n += 256) s += probs[(size_t)n * NEXP + e];
    red[threadIdx.x] = s;
    __syncthreads();
    for (int o = 128; o; o >>= 1) {
        if ((int)threadIdx.x < o) red[threadIdx.x] += red[threadIdx.x + o];
        __syncthreads();
    }
    if (threadIdx.x == 0) counts[e] = red[0];
}

__global__ void aux_kernel(const float* __restrict__ counts,
                           float* __restrict__ auxout) {
    if (threadIdx.x == 0 && blockIdx.x == 0) {
        float tot = 0.f;
        for (int e = 0; e < NEXP; ++e) tot += counts[e];
        float s = 0.f;
        for (int e = 0; e < NEXP; ++e)
            s += (counts[e] / tot) * (counts[e] / (float)N_TOK);
        auxout[0] = (float)NEXP * s;
    }
}

// ---------------- fused expert FFN ----------------
// One block = one expert, one tile of up to 16 routed (token,k) slots.
// 8 waves; kc-outer / tile-inner loop: A fragment loaded once per kc and
// reused across 8 independent WMMA accumulator chains (hazard-free issue).
// Layer-1 h chunk (16x1024 bf16) lives in LDS; layer-2 accumulators (16x128
// per wave) live in registers across all 4 EXP_DIM chunks.

__global__ __launch_bounds__(256)
void expert_ffn_kernel(const unsigned short* __restrict__ xb,
                       const unsigned short* __restrict__ Wt1,  // [E][F][H] bf16
                       const unsigned short* __restrict__ Wt2,  // [E][H][F] bf16
                       const float* __restrict__ b1,
                       const float* __restrict__ b2,
                       const float* __restrict__ wtop,
                       const int* __restrict__ list,
                       const int* __restrict__ cnt,
                       float* __restrict__ ybuf) {
    __shared__ __align__(16) unsigned short hs[16 * 1024];  // 32 KB h chunk
    __shared__ int   s_tok[16];
    __shared__ int   s_ent[16];
    __shared__ float s_w[16];

    const int e    = blockIdx.x >> 8;     // 256 tiles per expert
    const int tile = blockIdx.x & 255;
    const int mcnt = cnt[e];
    const int base = tile * 16;
    if (base >= mcnt) return;

    const int tid  = threadIdx.x;
    const int lane = tid & 31;
    const int wv   = tid >> 5;
    const int hi   = lane >> 4;       // 0 or 1
    const int nn   = lane & 15;       // row (A/C-M-half) or column index

    if (tid < 16) {
        int ent = (base + tid < mcnt) ? list[e * N_TOK + base + tid] : -1;
        s_ent[tid] = ent;
        s_tok[tid] = (ent >= 0) ? (ent >> 1) : 0;
        s_w[tid]   = (ent >= 0) ? wtop[ent] : 0.f;
    }
    __syncthreads();

    const int row_tok = s_tok[nn];

    v8f yacc[8];
#pragma unroll
    for (int nt = 0; nt < 8; ++nt) yacc[nt] = zero8();

    const unsigned short* aptr1 = xb + (size_t)row_tok * HID + hi * 8;

    for (int c = 0; c < 4; ++c) {                 // EXP_DIM in chunks of 1024
        // ---- layer 1: h[:, c*1024 .. +1023] ----
        v8f hacc[8];
#pragma unroll
        for (int ft = 0; ft < 8; ++ft) hacc[ft] = zero8();

        const unsigned short* bbase1 =
            Wt1 + ((size_t)(e * EXP_DIM + c * 1024 + wv * 128 + nn)) * HID + hi * 8;

        for (int kc = 0; kc < 32; ++kc) {         // K = HID
            FragU A;
            A.u[0] = *(const v4u*)(aptr1 + kc * 32);
            A.u[1] = *(const v4u*)(aptr1 + kc * 32 + 16);
#pragma unroll
            for (int ft = 0; ft < 8; ++ft) {
                FragU B;
                const unsigned short* bp = bbase1 + (size_t)ft * 16 * HID + kc * 32;
                B.u[0] = *(const v4u*)(bp);
                B.u[1] = *(const v4u*)(bp + 16);
                hacc[ft] = __builtin_amdgcn_wmma_f32_16x16x32_bf16(
                        false, A.v, false, B.v, (short)0, hacc[ft], false, false);
            }
        }
#pragma unroll
        for (int ft = 0; ft < 8; ++ft) {
            const int fcol_local = wv * 128 + ft * 16 + nn;
            const float bias = b1[e * EXP_DIM + c * 1024 + fcol_local];
#pragma unroll
            for (int v = 0; v < 8; ++v) {
                float hv = hacc[ft][v] + bias;
                hv = hv > 0.f ? hv : 0.f;
                hs[(v + 8 * hi) * 1024 + fcol_local] = f2bf(hv);
            }
        }
        __syncthreads();

        // ---- layer 2 partial accumulation over this chunk ----
        const unsigned short* bbase2 =
            Wt2 + ((size_t)(e * HID + wv * 128 + nn)) * EXP_DIM + c * 1024 + hi * 8;

        for (int kc = 0; kc < 32; ++kc) {
            FragU A;
            const unsigned short* ap = hs + nn * 1024 + kc * 32 + hi * 8;
            A.u[0] = *(const v4u*)(ap);
            A.u[1] = *(const v4u*)(ap + 16);
#pragma unroll
            for (int nt = 0; nt < 8; ++nt) {
                FragU B;
                const unsigned short* bp = bbase2 + (size_t)nt * 16 * EXP_DIM + kc * 32;
                B.u[0] = *(const v4u*)(bp);
                B.u[1] = *(const v4u*)(bp + 16);
                yacc[nt] = __builtin_amdgcn_wmma_f32_16x16x32_bf16(
                        false, A.v, false, B.v, (short)0, yacc[nt], false, false);
            }
        }
        __syncthreads();
    }

    // ---- epilogue: bias, gate-scale, scatter to per-(token,k) buffer ----
    float wreg[8];
    int   entv[8];
#pragma unroll
    for (int v = 0; v < 8; ++v) {
        int m = v + 8 * hi;
        wreg[v] = s_w[m];
        entv[v] = s_ent[m];
    }
#pragma unroll
    for (int nt = 0; nt < 8; ++nt) {
        const int ncol = wv * 128 + nt * 16 + nn;
        const float bias = b2[e * HID + ncol];
#pragma unroll
        for (int v = 0; v < 8; ++v) {
            if (entv[v] >= 0)
                ybuf[(size_t)entv[v] * HID + ncol] = (yacc[nt][v] + bias) * wreg[v];
        }
    }
}

// deterministic combine of the two expert contributions per token
__global__ void combine_kernel(const float* __restrict__ ybuf,
                               float* __restrict__ out, int n) {
    int i = blockIdx.x * 256 + threadIdx.x;
    if (i < n) {
        int tok = i / HID;
        size_t b = (size_t)i + (size_t)tok * HID;   // = (2*tok)*HID + d
        out[i] = ybuf[b] + ybuf[b + HID];
    }
}

// ---------------- launch ----------------

extern "C" void kernel_launch(void* const* d_in, const int* in_sizes, int n_in,
                              void* d_out, int out_size, void* d_ws, size_t ws_size,
                              hipStream_t stream) {
    (void)in_sizes; (void)n_in; (void)out_size; (void)ws_size;

    const float* x  = (const float*)d_in[0];
    const float* Wg = (const float*)d_in[1];
    const float* bg = (const float*)d_in[2];
    const float* W1 = (const float*)d_in[3];
    const float* b1 = (const float*)d_in[4];
    const float* W2 = (const float*)d_in[5];
    const float* b2 = (const float*)d_in[6];
    float* out = (float*)d_out;

    char* ws = (char*)d_ws;
    float*          counts = (float*)(ws + 0);
    int*            cnt    = (int*)(ws + 32);
    float*          wtop   = (float*)(ws + 256);                    // 8192 f
    int*            list   = (int*)(ws + 33024);                    // 8*4096 i
    float*          probs  = (float*)(ws + 164096);                 // 4096*8 f
    unsigned short* xb     = (unsigned short*)(ws + 295168);        // 4M bf16
    unsigned short* Wt1    = (unsigned short*)(ws + 8683776);       // 32M bf16
    unsigned short* Wt2    = (unsigned short*)(ws + 75792640);      // 32M bf16
    float*          ybuf   = (float*)(ws + 142901504);              // 8192*1024 f

    hipMemsetAsync(ws, 0, 256, stream);

    pack_x_kernel<<<(N_TOK * HID) / 256, 256, 0, stream>>>(x, xb, N_TOK * HID);
    // W1: [E][HID][EXP_DIM] -> Wt1 [E][EXP_DIM][HID]
    pack_transpose_kernel<<<dim3(EXP_DIM / 32, HID / 32, NEXP), dim3(32, 8), 0, stream>>>(
        W1, Wt1, HID, EXP_DIM);
    // W2: [E][EXP_DIM][HID] -> Wt2 [E][HID][EXP_DIM]
    pack_transpose_kernel<<<dim3(HID / 32, EXP_DIM / 32, NEXP), dim3(32, 8), 0, stream>>>(
        W2, Wt2, EXP_DIM, HID);

    gate_kernel<<<N_TOK / 8, 256, 0, stream>>>(x, Wg, bg, probs, cnt, list, wtop);
    counts_kernel<<<NEXP, 256, 0, stream>>>(probs, counts);
    aux_kernel<<<1, 32, 0, stream>>>(counts, out + (size_t)N_TOK * HID);

    expert_ffn_kernel<<<NEXP * 256, 256, 0, stream>>>(
        xb, Wt1, Wt2, b1, b2, wtop, list, cnt, ybuf);

    combine_kernel<<<(N_TOK * HID) / 256, 256, 0, stream>>>(ybuf, out, N_TOK * HID);
}
